// GATLayer_66511863545923
// MI455X (gfx1250) — matrix operand order repaired and checked by reading, not compiled, not run on hardware
//
#include <hip/hip_runtime.h>
#include <hip/hip_bf16.h>

typedef __attribute__((ext_vector_type(16))) _Float16 v16h;
typedef __attribute__((ext_vector_type(8)))  float    v8f;

#define N_NODES 8192
#define IN_F    128
#define OUT_F   64
#define NEG_INF -9000000000000000.0f

// ------------------------------------------------------------------
// Phase A: Wh = h @ W^T  (8192x64), f = Wh@a1, g = Wh@a2 (f32),
// and pack Wh into f16 WMMA B-fragments:
//   frag(kt, nt) covers K = kt*32..+31 (Wh rows), N = nt*16..+15 (Wh cols)
//   per-lane 16 contiguous halves; lanes 0-15: K=0..7,16..23 (2 per VGPR),
//   lanes 16-31: K=8..15,24..31 (CDNA5 ISA B layout).
// ------------------------------------------------------------------
__global__ __launch_bounds__(128) void gat_phaseA(
    const float* __restrict__ h, const float* __restrict__ W,
    const float* __restrict__ a, _Float16* __restrict__ whf16,
    float* __restrict__ fvec, float* __restrict__ gvec)
{
  __shared__ float Wl[OUT_F * IN_F];   // 32 KB
  __shared__ float hl[32 * IN_F];      // 16 KB
  __shared__ float whs[32 * 65];       // padded 32x64 Wh tile

  const int tid = threadIdx.x;
  const int i0  = blockIdx.x * 32;

  for (int i = tid; i < OUT_F * IN_F; i += 128) Wl[i] = W[i];
  for (int i = tid; i < 32 * IN_F;    i += 128) hl[i] = h[(size_t)i0 * IN_F + i];
  __syncthreads();

  for (int e = tid; e < 32 * OUT_F; e += 128) {
    int k = e >> 6, n = e & 63;
    const float* hp = &hl[k * IN_F];
    const float* wp = &Wl[n * IN_F];
    float acc = 0.f;
    for (int c = 0; c < IN_F; ++c) acc = fmaf(hp[c], wp[c], acc);
    whs[k * 65 + n] = acc;
  }
  __syncthreads();

  // B-fragment pack: wave = nt
  {
    const int wave = tid >> 5, lane = tid & 31;
    const int n  = wave * 16 + (lane & 15);
    const int hi = lane >> 4;
    v16h fr;
    for (int j = 0; j < 8; ++j) {
      int kb = hi * 8 + 2 * (j & 3) + 16 * (j >> 2);
      fr[2 * j]     = (_Float16)whs[kb * 65 + n];
      fr[2 * j + 1] = (_Float16)whs[(kb + 1) * 65 + n];
    }
    *(v16h*)(whf16 + ((size_t)(blockIdx.x * 4 + wave) * 32 + lane) * 16) = fr;
  }
  // f, g (full f32 precision)
  if (tid < 64) {
    const int r = tid & 31;
    const float* av = (tid < 32) ? a : a + OUT_F;
    float acc = 0.f;
    for (int n = 0; n < OUT_F; ++n) acc = fmaf(whs[r * 65 + n], av[n], acc);
    if (tid < 32) fvec[i0 + r] = acc; else gvec[i0 + r] = acc;
  }
}

// ------------------------------------------------------------------
// Phase B: per block = 16 rows. Pass 1: single adj read -> LDS bitmask +
// online softmax stats. Pass 2: recompute scores from bitmask, write
// attention (f32), feed f16 A-fragments into v_wmma_f32_16x16x32_f16
// against pre-packed Wh B-fragments; f32 C accumulation; cross-wave
// reduce -> h_prime.
// ------------------------------------------------------------------
__global__ __launch_bounds__(128) void gat_phaseB(
    const int* __restrict__ adj, const _Float16* __restrict__ whf16,
    const float* __restrict__ fvec, const float* __restrict__ gvec,
    float* __restrict__ hprime, float* __restrict__ attn)
{
  __shared__ unsigned smask[16 * 257];  // adjacency bitmask, padded rows
  __shared__ float red[4096];           // pass1 (m,s) reduce / pass2 C reduce
  __shared__ float f_lds[16], m_row[16], inv_s[16];

  const int tid  = threadIdx.x;
  const int wave = tid >> 5;
  const int lane = tid & 31;
  const int row0 = blockIdx.x * 16;

  if (tid < 16) f_lds[tid] = fvec[row0 + tid];
  __syncthreads();

  // ---- pass 1: adj -> bitmask + online (max, sum-exp) per row ----
  for (int r = 0; r < 16; ++r) {
    const float  fr      = f_lds[r];
    const size_t rowbase = (size_t)(row0 + r) * N_NODES;
    float mt = -__builtin_inff(), st = 0.f;
    for (int c = 0; c < 64; ++c) {
      const int col = c * 128 + tid;
      const int av  = adj[rowbase + col];
      const float x = fr + gvec[col];
      const float e = x > 0.f ? x : 0.01f * x;
      const float sc = (av > 0) ? e : NEG_INF;
      const unsigned bal = __builtin_amdgcn_ballot_w32(av > 0);
      if (lane == 0) smask[r * 257 + c * 4 + wave] = bal;
      const float mn = fmaxf(mt, sc);
      st = st * __expf(mt - mn) + __expf(sc - mn);
      mt = mn;
    }
    red[r * 128 + tid]        = mt;
    red[2048 + r * 128 + tid] = st;
  }
  __syncthreads();
  if (tid < 16) {
    float m = -__builtin_inff(), s = 0.f;
    for (int t = 0; t < 128; ++t) {
      const float mt = red[tid * 128 + t];
      const float st = red[2048 + tid * 128 + t];
      const float mn = fmaxf(m, mt);
      s = s * __expf(m - mn) + st * __expf(mt - mn);
      m = mn;
    }
    m_row[tid] = m;
    inv_s[tid] = 1.f / s;
  }
  __syncthreads();

  // ---- pass 2: attention write + WMMA accumulation ----
  const int   m  = lane & 15;
  const int   hi = lane >> 4;
  const float fm = f_lds[m];
  const float mr = m_row[m];
  const float is = inv_s[m];
  float* attrow = attn + (size_t)(row0 + m) * N_NODES;

  v8f C0 = {}, C1 = {}, C2 = {}, C3 = {};
  for (int kt = wave; kt < N_NODES / 32; kt += 4) {
    const unsigned word = smask[m * 257 + kt];
    const int k0 = kt * 32;
    v16h A;
    // A-fragment K slots (ISA 16-bit A 16x32 layout):
    // k = (j%4)*2 + p + hi*8 + (j/4)*16
    for (int j = 0; j < 8; ++j) {
      const int kb = ((j & 3) << 1) + (hi << 3) + ((j >> 2) << 4);
      const int k  = k0 + kb;
      const float2 gg = *(const float2*)(gvec + k);
      const float x0 = fm + gg.x, x1 = fm + gg.y;
      const float e0 = x0 > 0.f ? x0 : 0.01f * x0;
      const float e1 = x1 > 0.f ? x1 : 0.01f * x1;
      const float s0 = ((word >> kb) & 1u) ? e0 : NEG_INF;
      const float s1 = ((word >> (kb + 1)) & 1u) ? e1 : NEG_INF;
      const float a0 = __expf(s0 - mr) * is;
      const float a1 = __expf(s1 - mr) * is;
      float2 stv; stv.x = a0; stv.y = a1;
      *(float2*)(attrow + k) = stv;          // each 16x32 chunk fully covers
      A[2 * j]     = (_Float16)a0;           // sixteen 128B lines -> L2 combines
      A[2 * j + 1] = (_Float16)a1;
    }
    const _Float16* bbase = whf16 + ((size_t)kt * 4 * 32 + lane) * 16;
    const v16h B0 = *(const v16h*)(bbase);
    const v16h B1 = *(const v16h*)(bbase + 32 * 16);
    const v16h B2 = *(const v16h*)(bbase + 64 * 16);
    const v16h B3 = *(const v16h*)(bbase + 96 * 16);
    C0 = __builtin_amdgcn_wmma_f32_16x16x32_f16(false, A, false, B0, (short)0, C0, false, false);
    C1 = __builtin_amdgcn_wmma_f32_16x16x32_f16(false, A, false, B1, (short)0, C1, false, false);
    C2 = __builtin_amdgcn_wmma_f32_16x16x32_f16(false, A, false, B2, (short)0, C2, false, false);
    C3 = __builtin_amdgcn_wmma_f32_16x16x32_f16(false, A, false, B3, (short)0, C3, false, false);
  }

  // ---- cross-wave C reduction -> h_prime ----
  for (int v = 0; v < 8; ++v) {
    red[((wave * 4 + 0) * 8 + v) * 32 + lane] = C0[v];
    red[((wave * 4 + 1) * 8 + v) * 32 + lane] = C1[v];
    red[((wave * 4 + 2) * 8 + v) * 32 + lane] = C2[v];
    red[((wave * 4 + 3) * 8 + v) * 32 + lane] = C3[v];
  }
  __syncthreads();
  // wave w owns nt = w; D layout: M = v + 8*hi, N = lane%16
  for (int v = 0; v < 8; ++v) {
    const float s = red[((0 * 4 + wave) * 8 + v) * 32 + lane]
                  + red[((1 * 4 + wave) * 8 + v) * 32 + lane]
                  + red[((2 * 4 + wave) * 8 + v) * 32 + lane]
                  + red[((3 * 4 + wave) * 8 + v) * 32 + lane];
    hprime[(size_t)(row0 + v + hi * 8) * OUT_F + wave * 16 + m] = s;
  }
}

extern "C" void kernel_launch(void* const* d_in, const int* in_sizes, int n_in,
                              void* d_out, int out_size, void* d_ws, size_t ws_size,
                              hipStream_t stream) {
  (void)in_sizes; (void)n_in; (void)out_size; (void)ws_size;
  const float* h   = (const float*)d_in[0];
  const int*   adj = (const int*)d_in[1];
  const float* W   = (const float*)d_in[2];
  const float* a   = (const float*)d_in[3];

  float* out    = (float*)d_out;
  float* hprime = out;                                   // 8192*64
  float* attn   = out + (size_t)N_NODES * OUT_F;         // 8192*8192

  char* ws = (char*)d_ws;
  _Float16* whf16 = (_Float16*)ws;                       // 1 MB f16 B-fragments
  float* fvec = (float*)(ws + (size_t)N_NODES * OUT_F * sizeof(_Float16));
  float* gvec = fvec + N_NODES;

  gat_phaseA<<<N_NODES / 32, 128, 0, stream>>>(h, W, a, whf16, fvec, gvec);
  gat_phaseB<<<N_NODES / 16, 128, 0, stream>>>(adj, whf16, fvec, gvec, hprime, attn);
}
